// VarianceAdaptor_39968965656969
// MI455X (gfx1250) — matrix operand order repaired and checked by reading, not compiled, not run on hardware
//
#include <hip/hip_runtime.h>

// ---------------------------------------------------------------------------
// VarianceAdaptor for MI455X (gfx1250): bf16 WMMA GEMM convs + fused epilogues
// Round 2: double-buffered (software-pipelined) LDS staging + packed b64
// LDS stores for both A and B tiles.
// ---------------------------------------------------------------------------

typedef __attribute__((ext_vector_type(16))) __bf16 v16bf;
typedef __attribute__((ext_vector_type(8)))  __bf16 v8bf;
typedef __attribute__((ext_vector_type(8)))  float  v8f;

#define BB   32
#define TT   512
#define HH   512      // H == F == per-tap K
#define MM   (BB*TT)  // 16384 rows
#define TAPS 3
#define MAXL 2048
#define NBNS 255

// LDS tile row stride (bf16 elements); 40*2 = 80 bytes => 8/16B-aligned chunks
#define LDA 40
#define KSTEPS (TAPS * (HH / 32))   // 48 pipeline iterations

// ===========================================================================
// Tiled conv-as-GEMM:  C[M,512] = sum_tap A_shift(tap)[M,512] x W[tap][512,512]
// Block: 256 threads (8 waves). Tile: 128(M) x 128(N), K-step 32.
// Wave grid 4(M)x2(N): each wave -> 2x4 WMMA 16x16x32 bf16 accumulators.
// Double-buffered LDS: global loads for step it+1 overlap the WMMAs of step it.
// ===========================================================================
__global__ __launch_bounds__(256)
void conv_gemm_wmma(const float* __restrict__ A,       // [M, 512]
                    const float* __restrict__ W,       // [3, 512, 512] tap-major
                    float* __restrict__ C,             // [M, 512]
                    const int*   __restrict__ emb_idx, // [M] or null
                    const float* __restrict__ emb_tab) // [256, 512] or null
{
    __shared__ __align__(16) __bf16 ldsA[2][128][LDA];
    __shared__ __align__(16) __bf16 ldsB[2][128][LDA];   // transposed: [n][k]

    const int tid  = threadIdx.x;
    const int lane = tid & 31;
    const int wid  = tid >> 5;
    const int wm   = wid & 3;          // 32-row slab within tile
    const int wn   = wid >> 2;         // 64-col slab within tile
    const int m_base = blockIdx.y * 128;
    const int n_base = blockIdx.x * 128;

    v8f acc[2][4];
    #pragma unroll
    for (int ms = 0; ms < 2; ++ms)
        #pragma unroll
        for (int ns = 0; ns < 4; ++ns)
            #pragma unroll
            for (int i = 0; i < 8; ++i)
                acc[ms][ns][i] = 0.0f;

    const int lrow  = lane & 15;
    const int ahalf = (lane < 16) ? 0 : 8;    // A: K 0-7/16-23 vs 8-15/24-31
    const int bhalf = (lane < 16) ? 0 : 8;    // B: K 0-7/16-23 vs 8-15/24-31

    // ---- per-thread staging geometry (loop-invariant) ---------------------
    // A: 4 chunks of 4 floats: chunk j covers row (j*256+tid)>>3, cols ((..)&7)*4
    // B: one 4(k) x 4(n) sub-block: rows rB..rB+3, cols cB..cB+3
    const int cB = (tid & 31) * 4;
    const int rB = (tid >> 5) * 4;

    float4 a_reg[4];
    float4 b_reg[4];

    // stage loader: global -> registers (with im2col shift + optional emb add)
    auto load_tiles = [&](int it) {
        const int tap = it >> 4;
        const int kk  = (it & 15) * 32;
        #pragma unroll
        for (int j = 0; j < 4; ++j) {
            int q   = j * 256 + tid;
            int row = q >> 3;
            int c4  = (q & 7) * 4;
            int rg  = m_base + row;
            int b   = rg >> 9;
            int t   = rg & (TT - 1);
            int ts  = t + tap - 1;        // pad = 1
            float4 v = make_float4(0.f, 0.f, 0.f, 0.f);
            if ((unsigned)ts < (unsigned)TT) {
                int arow = (b << 9) | ts;
                v = *(const float4*)(A + (size_t)arow * HH + kk + c4);
                if (emb_idx) {
                    const float4 e = *(const float4*)(emb_tab +
                                     (size_t)emb_idx[arow] * HH + kk + c4);
                    v.x += e.x; v.y += e.y; v.z += e.z; v.w += e.w;
                }
            }
            a_reg[j] = v;
        }
        const float* Wt = W + (size_t)tap * HH * HH;
        #pragma unroll
        for (int i = 0; i < 4; ++i)
            b_reg[i] = *(const float4*)(Wt + (size_t)(kk + rB + i) * HH + n_base + cB);
    };

    // stage storer: registers -> LDS buffer (packed 8-byte bf16 stores)
    auto store_tiles = [&](int buf) {
        #pragma unroll
        for (int j = 0; j < 4; ++j) {
            int q   = j * 256 + tid;
            int row = q >> 3;
            int c4  = (q & 7) * 4;
            union { __bf16 h[4]; uint2 u; } pk;
            pk.h[0] = (__bf16)a_reg[j].x; pk.h[1] = (__bf16)a_reg[j].y;
            pk.h[2] = (__bf16)a_reg[j].z; pk.h[3] = (__bf16)a_reg[j].w;
            *(uint2*)&ldsA[buf][row][c4] = pk.u;        // 8B aligned: c4%4==0
        }
        // B transpose in registers: column jn gets k-values rB..rB+3
        #pragma unroll
        for (int jn = 0; jn < 4; ++jn) {
            union { __bf16 h[4]; uint2 u; } pk;
            pk.h[0] = (__bf16)((const float*)&b_reg[0])[jn];
            pk.h[1] = (__bf16)((const float*)&b_reg[1])[jn];
            pk.h[2] = (__bf16)((const float*)&b_reg[2])[jn];
            pk.h[3] = (__bf16)((const float*)&b_reg[3])[jn];
            *(uint2*)&ldsB[buf][cB + jn][rB] = pk.u;    // 8B aligned: rB%4==0
        }
    };

    // ---- pipeline prologue ------------------------------------------------
    load_tiles(0);
    store_tiles(0);
    __syncthreads();

    // ---- main pipelined K loop -------------------------------------------
    for (int it = 0; it < KSTEPS; ++it) {
        const int cur = it & 1;
        const bool has_next = (it + 1) < KSTEPS;
        if (has_next) load_tiles(it + 1);   // overlap global loads with WMMA

        v16bf afrag[2];
        #pragma unroll
        for (int ms = 0; ms < 2; ++ms) {
            int row = wm * 32 + ms * 16 + lrow;
            union { v16bf v; v8bf h[2]; } u;
            u.h[0] = *(const v8bf*)&ldsA[cur][row][ahalf];
            u.h[1] = *(const v8bf*)&ldsA[cur][row][ahalf + 16];
            afrag[ms] = u.v;
        }
        #pragma unroll
        for (int ns = 0; ns < 4; ++ns) {
            int col = wn * 64 + ns * 16 + lrow;
            union { v16bf v; v8bf h[2]; } u;
            u.h[0] = *(const v8bf*)&ldsB[cur][col][bhalf];
            u.h[1] = *(const v8bf*)&ldsB[cur][col][bhalf + 16];
            v16bf bfrag = u.v;
            #pragma unroll
            for (int ms = 0; ms < 2; ++ms)
                acc[ms][ns] = __builtin_amdgcn_wmma_f32_16x16x32_bf16(
                    false, afrag[ms], false, bfrag,
                    (short)0, acc[ms][ns], false, false);
        }

        if (has_next) store_tiles(cur ^ 1);
        __syncthreads();
    }

    // ---- write back C tile (f32). C/D layout: lane%16 = col, lane/16 -> M+8.
    const int rofs = (lane >> 4) << 3;
    const int cofs = lane & 15;
    #pragma unroll
    for (int ms = 0; ms < 2; ++ms)
        #pragma unroll
        for (int ns = 0; ns < 4; ++ns)
            #pragma unroll
            for (int v = 0; v < 8; ++v) {
                int r = m_base + wm * 32 + ms * 16 + v + rofs;
                int c = n_base + wn * 64 + ns * 16 + cofs;
                C[(size_t)r * HH + c] = acc[ms][ns][v];
            }
}

// ===========================================================================
// bias + ReLU + LayerNorm over F=512, fused optional Linear(F->1)+mask,
// or feature write for the next conv. One 256-thread block per row.
// ===========================================================================
__global__ __launch_bounds__(256)
void ln_epilogue(const float* __restrict__ Y, const float* __restrict__ bias,
                 const float* __restrict__ gamma, const float* __restrict__ beta,
                 float* __restrict__ out_feat,           // [M,512] or null
                 const float* __restrict__ lw,           // [512]   or null
                 const float* __restrict__ lb,           // [1]     or null
                 const unsigned char* __restrict__ mask, // [M]
                 float* __restrict__ out_scalar)         // [M] or null
{
    const int row = blockIdx.x;
    const int tid = threadIdx.x;
    __shared__ float s_sum[256];
    __shared__ float s_sq[256];

    float v0 = fmaxf(Y[(size_t)row * HH + tid]       + bias[tid],       0.f);
    float v1 = fmaxf(Y[(size_t)row * HH + 256 + tid] + bias[256 + tid], 0.f);
    s_sum[tid] = v0 + v1;
    s_sq[tid]  = v0 * v0 + v1 * v1;
    __syncthreads();
    #pragma unroll
    for (int ofs = 128; ofs > 0; ofs >>= 1) {
        if (tid < ofs) { s_sum[tid] += s_sum[tid + ofs]; s_sq[tid] += s_sq[tid + ofs]; }
        __syncthreads();
    }
    const float mean = s_sum[0] * (1.0f / 512.0f);
    const float var  = s_sq[0] * (1.0f / 512.0f) - mean * mean;
    const float inv  = rsqrtf(var + 1e-5f);
    const float n0 = (v0 - mean) * inv * gamma[tid]       + beta[tid];
    const float n1 = (v1 - mean) * inv * gamma[256 + tid] + beta[256 + tid];

    if (out_feat) {
        out_feat[(size_t)row * HH + tid]       = n0;
        out_feat[(size_t)row * HH + 256 + tid] = n1;
    }
    if (out_scalar) {
        float p = n0 * lw[tid] + n1 * lw[256 + tid];
        __syncthreads();
        s_sum[tid] = p;
        __syncthreads();
        #pragma unroll
        for (int ofs = 128; ofs > 0; ofs >>= 1) {
            if (tid < ofs) s_sum[tid] += s_sum[tid + ofs];
            __syncthreads();
        }
        if (tid == 0)
            out_scalar[row] = mask[row] ? 0.f : (s_sum[0] + lb[0]);
    }
}

// ===========================================================================
// idx = searchsorted(bins, v, side='left') = count(bins[j] < v)
// ===========================================================================
__global__ void bucketize_kernel(const float* __restrict__ bins,
                                 const float* __restrict__ tgt,
                                 int* __restrict__ idx, int n)
{
    int i = blockIdx.x * blockDim.x + threadIdx.x;
    if (i >= n) return;
    float v = tgt[i];
    int c = 0;
    for (int j = 0; j < NBNS; ++j) c += (bins[j] < v) ? 1 : 0;
    idx[i] = c;
}

// ===========================================================================
// Masked-duration inclusive scan per batch row + mel_len.
// ===========================================================================
__global__ __launch_bounds__(512)
void cumsum_kernel(const int* __restrict__ duration,
                   const unsigned char* __restrict__ mask,
                   const int* __restrict__ max_len_p,
                   int* __restrict__ cum, float* __restrict__ mel_len_out)
{
    const int b = blockIdx.x, t = threadIdx.x;
    __shared__ int s[512];
    s[t] = mask[b * TT + t] ? 0 : duration[b * TT + t];
    __syncthreads();
    for (int ofs = 1; ofs < 512; ofs <<= 1) {
        int add = (t >= ofs) ? s[t - ofs] : 0;
        __syncthreads();
        s[t] += add;
        __syncthreads();
    }
    cum[b * TT + t] = s[t];
    if (t == 511) mel_len_out[b] = (float)min(s[511], max_len_p[0]);
}

// ===========================================================================
// Length regulator: one block per (b, frame). Recomputes the embedding adds
// at gather time (x + pitch_emb + energy_emb), so no f32 copy of x2/x3 exists.
// ===========================================================================
__global__ __launch_bounds__(256)
void length_regulate(const float* __restrict__ x,
                     const int* __restrict__ p_idx, const float* __restrict__ p_tab,
                     const int* __restrict__ e_idx, const float* __restrict__ e_tab,
                     const int* __restrict__ cum, const int* __restrict__ max_len_p,
                     float* __restrict__ x_out, float* __restrict__ mel_mask)
{
    const int m = blockIdx.x;   // frame 0..MAXL-1
    const int b = blockIdx.y;
    const int* c = cum + b * TT;
    const int ml = min(c[TT - 1], max_len_p[0]);
    const bool msk = (m >= ml);

    // searchsorted(c, m, 'right'): first i with c[i] > m  (uniform binary search)
    int lo = 0, hi = TT;
    while (lo < hi) { int mid = (lo + hi) >> 1; if (c[mid] <= m) lo = mid + 1; else hi = mid; }
    const int idx = min(lo, TT - 1);

    const int tid = threadIdx.x;
    const int r = b * TT + idx;
    const int pi = p_idx[r], ei = e_idx[r];
    for (int f = tid; f < HH; f += 256) {
        float v = 0.f;
        if (!msk)
            v = x[(size_t)r * HH + f] + p_tab[(size_t)pi * HH + f]
                                      + e_tab[(size_t)ei * HH + f];
        x_out[((size_t)b * MAXL + m) * HH + f] = v;
    }
    if (tid == 0) mel_mask[(size_t)b * MAXL + m] = msk ? 1.f : 0.f;
}

// ===========================================================================
extern "C" void kernel_launch(void* const* d_in, const int* in_sizes, int n_in,
                              void* d_out, int out_size, void* d_ws, size_t ws_size,
                              hipStream_t stream)
{
    const float* x            = (const float*)d_in[0];
    const unsigned char* smask= (const unsigned char*)d_in[1];
    const int*   duration     = (const int*)d_in[2];
    const float* pitch_target = (const float*)d_in[3];
    const float* energy_target= (const float*)d_in[4];
    const float* conv1_w      = (const float*)d_in[5];
    const float* conv1_b      = (const float*)d_in[6];
    const float* ln1_g        = (const float*)d_in[7];
    const float* ln1_b        = (const float*)d_in[8];
    const float* conv2_w      = (const float*)d_in[9];
    const float* conv2_b      = (const float*)d_in[10];
    const float* ln2_g        = (const float*)d_in[11];
    const float* ln2_b        = (const float*)d_in[12];
    const float* lin_w        = (const float*)d_in[13];
    const float* lin_b        = (const float*)d_in[14];
    const float* pitch_bins   = (const float*)d_in[15];
    const float* energy_bins  = (const float*)d_in[16];
    const float* pitch_table  = (const float*)d_in[17];
    const float* energy_table = (const float*)d_in[18];
    const int*   max_len      = (const int*)d_in[19];

    float* out          = (float*)d_out;
    float* x_out        = out;                                  // [32,2048,512]
    float* logdur       = out + (size_t)BB * MAXL * HH;         // [32,512]
    float* pitchp       = logdur + BB * TT;
    float* energyp      = pitchp + BB * TT;
    float* mel_len_out  = energyp + BB * TT;                    // [32]
    float* mel_mask_out = mel_len_out + BB;                     // [32,2048]

    char* ws = (char*)d_ws;
    const size_t MF = (size_t)MM * HH;
    float* tmp1 = (float*)ws;  ws += MF * sizeof(float);
    float* tmp2 = (float*)ws;  ws += MF * sizeof(float);
    int* p_idx  = (int*)ws;    ws += MM * sizeof(int);
    int* e_idx  = (int*)ws;    ws += MM * sizeof(int);
    int* cum    = (int*)ws;    ws += MM * sizeof(int);

    const dim3 ggrid(HH / 128, MM / 128);   // 4 x 128
    const dim3 gblk(256);
    const size_t WS = (size_t)TAPS * HH * HH;   // per-predictor conv weight stride

    bucketize_kernel<<<(MM + 255) / 256, 256, 0, stream>>>(pitch_bins, pitch_target, p_idx, MM);
    bucketize_kernel<<<(MM + 255) / 256, 256, 0, stream>>>(energy_bins, energy_target, e_idx, MM);

    // ---- predictor 0: duration (input x)
    conv_gemm_wmma<<<ggrid, gblk, 0, stream>>>(x, conv1_w + 0 * WS, tmp1, nullptr, nullptr);
    ln_epilogue<<<MM, 256, 0, stream>>>(tmp1, conv1_b, ln1_g, ln1_b,
                                        tmp2, nullptr, nullptr, smask, nullptr);
    conv_gemm_wmma<<<ggrid, gblk, 0, stream>>>(tmp2, conv2_w + 0 * WS, tmp1, nullptr, nullptr);
    ln_epilogue<<<MM, 256, 0, stream>>>(tmp1, conv2_b, ln2_g, ln2_b,
                                        nullptr, lin_w, lin_b, smask, logdur);

    // ---- predictor 1: pitch (input x)
    conv_gemm_wmma<<<ggrid, gblk, 0, stream>>>(x, conv1_w + 1 * WS, tmp1, nullptr, nullptr);
    ln_epilogue<<<MM, 256, 0, stream>>>(tmp1, conv1_b + HH, ln1_g + HH, ln1_b + HH,
                                        tmp2, nullptr, nullptr, smask, nullptr);
    conv_gemm_wmma<<<ggrid, gblk, 0, stream>>>(tmp2, conv2_w + 1 * WS, tmp1, nullptr, nullptr);
    ln_epilogue<<<MM, 256, 0, stream>>>(tmp1, conv2_b + HH, ln2_g + HH, ln2_b + HH,
                                        nullptr, lin_w + HH, lin_b + 1, smask, pitchp);

    // ---- predictor 2: energy on (x + pitch_emb), embedding fused into A load
    conv_gemm_wmma<<<ggrid, gblk, 0, stream>>>(x, conv1_w + 2 * WS, tmp1, p_idx, pitch_table);
    ln_epilogue<<<MM, 256, 0, stream>>>(tmp1, conv1_b + 2 * HH, ln1_g + 2 * HH, ln1_b + 2 * HH,
                                        tmp2, nullptr, nullptr, smask, nullptr);
    conv_gemm_wmma<<<ggrid, gblk, 0, stream>>>(tmp2, conv2_w + 2 * WS, tmp1, nullptr, nullptr);
    ln_epilogue<<<MM, 256, 0, stream>>>(tmp1, conv2_b + 2 * HH, ln2_g + 2 * HH, ln2_b + 2 * HH,
                                        nullptr, lin_w + 2 * HH, lin_b + 2, smask, energyp);

    // ---- length regulator
    cumsum_kernel<<<BB, 512, 0, stream>>>(duration, smask, max_len, cum, mel_len_out);
    length_regulate<<<dim3(MAXL, BB), 256, 0, stream>>>(
        x, p_idx, pitch_table, e_idx, energy_table, cum, max_len, x_out, mel_mask_out);
}
